// TextConditionedDynamicLayerAttention_82471962018687
// MI455X (gfx1250) — compile-verified, matrix operand order checked by reading, not run on hardware
//
#include <hip/hip_runtime.h>
#include <math.h>

// ---------------------------------------------------------------------------
// Problem constants
// ---------------------------------------------------------------------------
#define DDIM   4096
#define LNUM   24
#define NPATCH 576
#define TTOK   128
#define REDDIM 1024
#define PTOT   (LNUM * NPATCH)      // 13824 patches
#define FOCUS  22                   // (-2) % 24

typedef __attribute__((ext_vector_type(8)))  float  v8f;
typedef __attribute__((ext_vector_type(16))) __bf16 v16bf;

union bfrag { unsigned int u[8]; v16bf v; };

__device__ __forceinline__ unsigned short f2bf(float x) {
    unsigned int u = __float_as_uint(x);
    u += 0x7FFFu + ((u >> 16) & 1u);       // round-to-nearest-even
    return (unsigned short)(u >> 16);
}

__device__ __forceinline__ unsigned int pack2(float a, float b) {
    return (unsigned int)f2bf(a) | ((unsigned int)f2bf(b) << 16);
}

__device__ __forceinline__ float sigm(float x) { return 1.0f / (1.0f + expf(-x)); }

template <int BS>
__device__ __forceinline__ float block_reduce(float v, float* red) {
    int t = threadIdx.x;
    __syncthreads();
    red[t] = v;
    __syncthreads();
#pragma unroll
    for (int o = BS / 2; o > 0; o >>= 1) {
        if (t < o) red[t] += red[t + o];
        __syncthreads();
    }
    return red[0];
}

// ---------------------------------------------------------------------------
// misc small kernels
// ---------------------------------------------------------------------------
__global__ void zero_kernel(float* p, int n) {
    int i = blockIdx.x * blockDim.x + threadIdx.x;
    if (i < n) p[i] = 0.0f;
}

// text_global = LN(mean_T(text));   1 block x 1024 threads, 4 cols/thread
__global__ void text_mean_ln_kernel(const float* __restrict__ text, float* __restrict__ tg) {
    __shared__ float red[1024];
    int t = threadIdx.x;
    float v[4];
    float ps = 0.0f, ps2 = 0.0f;
#pragma unroll
    for (int j = 0; j < 4; ++j) {
        int d = t + j * 1024;
        float su = 0.0f;
        for (int r = 0; r < TTOK; ++r) su += text[(size_t)r * DDIM + d];
        v[j] = su * (1.0f / (float)TTOK);
        ps += v[j];
        ps2 += v[j] * v[j];
    }
    float S  = block_reduce<1024>(ps,  red);
    float S2 = block_reduce<1024>(ps2, red);
    float mu  = S * (1.0f / (float)DDIM);
    float var = S2 * (1.0f / (float)DDIM) - mu * mu;
    float inv = 1.0f / sqrtf(var + 1e-5f);
#pragma unroll
    for (int j = 0; j < 4; ++j) {
        int d = t + j * 1024;
        tg[d] = (v[j] - mu) * inv;
    }
}

// y[l,d] = mean_n proj[l,n,d]; float4-wide; grid = L*D/4/256 = 96
__global__ void layer_mean_kernel(const float* __restrict__ proj, float* __restrict__ y) {
    int flat4 = blockIdx.x * 256 + threadIdx.x;       // float4 index within [L * D/4)
    int l  = flat4 / (DDIM / 4);
    int d4 = flat4 % (DDIM / 4);
    const float4* base = (const float4*)(proj + (size_t)l * NPATCH * DDIM) + d4;
    float sx = 0.f, sy = 0.f, sz = 0.f, sw = 0.f;
    for (int n = 0; n < NPATCH; ++n) {
        float4 v = base[(size_t)n * (DDIM / 4)];
        sx += v.x; sy += v.y; sz += v.z; sw += v.w;
    }
    const float inv = 1.0f / (float)NPATCH;
    ((float4*)y)[flat4] = make_float4(sx * inv, sy * inv, sz * inv, sw * inv);
}

// s[r] = relu(W1[r,:] . [sigm(c), y_l, tg] + b1[r]);  grid = RED blocks x 128
__global__ void dsu_s_kernel(const float* __restrict__ W1w, const float* __restrict__ W1b,
                             const float* __restrict__ c,   const float* __restrict__ yl,
                             const float* __restrict__ tg,  float* __restrict__ s) {
    __shared__ float red[128];
    int r = blockIdx.x;
    const float* row = W1w + (size_t)r * (3 * DDIM);
    float p = 0.0f;
    for (int j = threadIdx.x; j < 3 * DDIM; j += 128) {
        float v;
        if (j < DDIM)          v = sigm(c[j]);
        else if (j < 2 * DDIM) v = yl[j - DDIM];
        else                   v = tg[j - 2 * DDIM];
        p += row[j] * v;
    }
    float tot = block_reduce<128>(p, red);
    if (threadIdx.x == 0) s[r] = fmaxf(tot + W1b[r], 0.0f);
}

// c[d] = f*c[d] + i*tanh(...);  grid = D blocks x 128
__global__ void dsu_c_kernel(const float* __restrict__ Wcw, const float* __restrict__ Wcb, const float* __restrict__ bcv,
                             const float* __restrict__ Wiw, const float* __restrict__ Wib, const float* __restrict__ biv,
                             const float* __restrict__ Wfw, const float* __restrict__ Wfb, const float* __restrict__ bfv,
                             const float* __restrict__ s, float* __restrict__ c,
                             float* __restrict__ cfinal, int save) {
    __shared__ float red[128];
    int d = blockIdx.x;
    const float* rc = Wcw + (size_t)d * REDDIM;
    const float* ri = Wiw + (size_t)d * REDDIM;
    const float* rf = Wfw + (size_t)d * REDDIM;
    float pc = 0.0f, pi = 0.0f, pf = 0.0f;
    for (int r = threadIdx.x; r < REDDIM; r += 128) {
        float sv = s[r];
        pc += rc[r] * sv; pi += ri[r] * sv; pf += rf[r] * sv;
    }
    float dc = block_reduce<128>(pc, red);
    float di = block_reduce<128>(pi, red);
    float df = block_reduce<128>(pf, red);
    if (threadIdx.x == 0) {
        float ct = tanhf(dc + Wcb[d] + bcv[d]);
        float ig = sigm(di + Wib[d] + biv[d]);
        float fg = sigm(df + Wfb[d] + bfv[d]);
        float cn = fg * c[d] + ig * ct;
        c[d] = cn;
        if (save) cfinal[d] = cn;
    }
}

// qraw[d] = Wq[d,:] . c_final;  grid = D blocks x 256
__global__ void qmatvec_kernel(const float* __restrict__ Wq, const float* __restrict__ cf,
                               float* __restrict__ qraw) {
    __shared__ float red[256];
    int d = blockIdx.x;
    const float* row = Wq + (size_t)d * DDIM;
    float p = 0.0f;
    for (int e = threadIdx.x; e < DDIM; e += 256) p += row[e] * cf[e];
    float tot = block_reduce<256>(p, red);
    if (threadIdx.x == 0) qraw[d] = tot;
}

// q = LN(qraw)*ln_w + ln_b ; wq = ln_w*q ; consts = {sum(wq), sum(ln_b*q)}
__global__ void q_ln_kernel(const float* __restrict__ qraw,
                            const float* __restrict__ lnw, const float* __restrict__ lnb,
                            float* __restrict__ q, float* __restrict__ wq,
                            float* __restrict__ consts) {
    __shared__ float red[1024];
    int t = threadIdx.x;
    float v[4];
    float ps = 0.0f, ps2 = 0.0f;
#pragma unroll
    for (int j = 0; j < 4; ++j) {
        int d = t + j * 1024;
        v[j] = qraw[d];
        ps += v[j]; ps2 += v[j] * v[j];
    }
    float S  = block_reduce<1024>(ps,  red);
    float S2 = block_reduce<1024>(ps2, red);
    float mu  = S * (1.0f / (float)DDIM);
    float var = S2 * (1.0f / (float)DDIM) - mu * mu;
    float inv = 1.0f / sqrtf(var + 1e-5f);
    float pB = 0.0f, pC = 0.0f;
#pragma unroll
    for (int j = 0; j < 4; ++j) {
        int d = t + j * 1024;
        float qq = (v[j] - mu) * inv * lnw[d] + lnb[d];
        q[d]  = qq;
        float w = lnw[d] * qq;
        wq[d] = w;
        pB += w;
        pC += lnb[d] * qq;
    }
    float B  = block_reduce<1024>(pB, red);
    float Cc = block_reduce<1024>(pC, red);
    if (t == 0) { consts[0] = B; consts[1] = Cc; }
}

// ---------------------------------------------------------------------------
// WMMA score kernel: kraw = proj @ Wk.T (bf16 WMMA, K=4096) fused with
// LayerNorm statistics + q-dot so kraw is never written to memory.
// grid = 216 blocks (64 patches each) x 256 threads (8 waves).
// Staging: float4 global loads -> packed bf16 -> uint2 LDS stores.
// K-chunk = 64 (two WMMA K-substeps per LDS stage).
// ---------------------------------------------------------------------------
__global__ void __launch_bounds__(256)
score_wmma_kernel(const float* __restrict__ proj, const float* __restrict__ Wk,
                  const float* __restrict__ wq, const float* __restrict__ consts,
                  float* __restrict__ scores) {
    __shared__ unsigned int sA[64 * 32];      // proj chunk: 64 rows x 64 bf16 (32 dw/row)
    __shared__ unsigned int sB[128 * 32];     // Wk   chunk: 128 rows x 64 bf16
    __shared__ float aS1[8][64], aS2[8][64], aA[8][64];

    const int tid  = threadIdx.x;
    const int wave = tid >> 5;
    const int lane = tid & 31;
    const int halfsel = lane >> 4;            // 0: lanes 0-15, 1: lanes 16-31
    const int p0 = halfsel ? 4 : 0;           // dword picks per ISA 16-bit A/B layout
    const int p1 = halfsel ? 12 : 8;
    const int mbase = blockIdx.x * 64;

    if (tid < 64) {
        for (int w = 0; w < 8; ++w) { aS1[w][tid] = 0.f; aS2[w][tid] = 0.f; aA[w][tid] = 0.f; }
    }
    __syncthreads();

    for (int ep = 0; ep < 32; ++ep) {                      // 128 output cols / pass
        const int ebase = ep * 128;
        const float wql = wq[ebase + (wave << 4) + (lane & 15)];
        v8f acc[4] = {};
        for (int ks = 0; ks < 64; ++ks) {                  // K chunks of 64
            const int kbase = ks * 64;
            __syncthreads();                               // WAR vs previous frag reads
#pragma unroll
            for (int j = 0; j < 4; ++j) {                  // stage proj 64x64: 1024 float4
                int flat4 = tid + j * 256;
                int m = flat4 >> 4, kq = flat4 & 15;
                float4 v = *((const float4*)(proj + (size_t)(mbase + m) * DDIM + kbase) + kq);
                ((uint2*)sA)[flat4] = make_uint2(pack2(v.x, v.y), pack2(v.z, v.w));
            }
#pragma unroll
            for (int j = 0; j < 8; ++j) {                  // stage Wk 128x64: 2048 float4
                int flat4 = tid + j * 256;
                int e = flat4 >> 4, kq = flat4 & 15;
                float4 v = *((const float4*)(Wk + (size_t)(ebase + e) * DDIM + kbase) + kq);
                ((uint2*)sB)[flat4] = make_uint2(pack2(v.x, v.y), pack2(v.z, v.w));
            }
            if (ks < 63) {                                 // prefetch next K chunk of Wk
                __builtin_prefetch((const void*)(Wk + (size_t)(ebase + (tid >> 1)) * DDIM + kbase + 64), 0, 1);
            }
            __syncthreads();

#pragma unroll
            for (int h = 0; h < 2; ++h) {                  // two K=32 substeps
                const int q0 = h * 16 + p0;
                const int q1 = h * 16 + p1;
                bfrag b;
                {
                    const unsigned int* row = sB + ((wave << 4) + (lane & 15)) * 32;
#pragma unroll
                    for (int i = 0; i < 4; ++i) { b.u[i] = row[q0 + i]; b.u[4 + i] = row[q1 + i]; }
                }
#pragma unroll
                for (int mt = 0; mt < 4; ++mt) {
                    bfrag a;
                    const unsigned int* row = sA + ((mt << 4) + (lane & 15)) * 32;
#pragma unroll
                    for (int i = 0; i < 4; ++i) { a.u[i] = row[q0 + i]; a.u[4 + i] = row[q1 + i]; }
                    acc[mt] = __builtin_amdgcn_wmma_f32_16x16x32_bf16(
                        false, a.v, false, b.v, (short)0, acc[mt], false, false);
                }
            }
        }
        // fold this 64x128 kraw tile into per-patch LN/score partials
#pragma unroll
        for (int mt = 0; mt < 4; ++mt) {
#pragma unroll
            for (int r = 0; r < 8; ++r) {
                float kv = acc[mt][r];
                float s1 = kv, s2 = kv * kv, aa = kv * wql;
#pragma unroll
                for (int off = 8; off >= 1; off >>= 1) {   // reduce 16-lane halves
                    s1 += __shfl_xor(s1, off, 32);
                    s2 += __shfl_xor(s2, off, 32);
                    aa += __shfl_xor(aa, off, 32);
                }
                if ((lane & 15) == 0) {
                    int m = (mt << 4) + r + (halfsel ? 8 : 0);
                    aS1[wave][m] += s1; aS2[wave][m] += s2; aA[wave][m] += aa;
                }
            }
        }
    }
    __syncthreads();
    if (tid < 64) {
        float S1 = 0.f, S2 = 0.f, A = 0.f;
        for (int w = 0; w < 8; ++w) { S1 += aS1[w][tid]; S2 += aS2[w][tid]; A += aA[w][tid]; }
        float mu  = S1 * (1.0f / (float)DDIM);
        float var = S2 * (1.0f / (float)DDIM) - mu * mu;
        float sd  = sqrtf(var + 1e-5f);
        scores[mbase + tid] = (A - mu * consts[0]) / sd + consts[1];
    }
}

// per-layer z-normalization; grid = 24 blocks x 256
__global__ void znorm_kernel(const float* __restrict__ scores, float* __restrict__ z) {
    __shared__ float red[256];
    int l = blockIdx.x;
    const float* row = scores + (size_t)l * NPATCH;
    float p = 0.0f, p2 = 0.0f;
    for (int n = threadIdx.x; n < NPATCH; n += 256) {
        float x = row[n];
        p += x; p2 += x * x;
    }
    float S  = block_reduce<256>(p,  red);
    float S2 = block_reduce<256>(p2, red);
    float mu  = S * (1.0f / (float)NPATCH);
    float var = fmaxf(S2 * (1.0f / (float)NPATCH) - mu * mu, 0.0f);
    float sd  = sqrtf(var);
    for (int n = threadIdx.x; n < NPATCH; n += 256)
        z[(size_t)l * NPATCH + n] = (row[n] - mu) / (sd + 1e-6f);
}

// data-dependent selection, serial on one lane (tiny: 24x576 input, <=128 picks)
__global__ void select_kernel(const float* __restrict__ z,
                              int* __restrict__ sel_li, int* __restrict__ sel_idx) {
    if (threadIdx.x != 0) return;
    const float NEG = -3.0e38f;
    float conf[LNUM];
    for (int l = 0; l < LNUM; ++l) {
        float m = NEG;
        const float* row = z + (size_t)l * NPATCH;
        for (int n = 0; n < NPATCH; ++n) m = fmaxf(m, row[n]);
        conf[l] = m;
    }
    int kper[LNUM];
    for (int l = 0; l < LNUM; ++l) kper[l] = 0;
    kper[FOCUS] = 64;                                  // min(TOPK/2, N)
    int other[LNUM], no = 0;
    for (int l = 0; l < LNUM; ++l)
        if (l != FOCUS && conf[l] > 2.0f) other[no++] = l;
    int rest = 64;
    if (no > 0) {
        float w[LNUM]; float cmax = NEG;
        for (int j = 0; j < no; ++j) cmax = fmaxf(cmax, conf[other[j]]);
        float wsum = 0.0f;
        for (int j = 0; j < no; ++j) { w[j] = expf(conf[other[j]] - cmax); wsum += w[j]; }
        for (int j = 0; j < no; ++j) w[j] /= wsum;
        int alloc[LNUM]; int asum = 0;
        for (int j = 0; j < no; ++j) { alloc[j] = (int)floorf(w[j] * (float)rest); asum += alloc[j]; }
        int rem = rest - asum;
        bool taken[LNUM];
        for (int j = 0; j < no; ++j) taken[j] = false;
        while (rem > 0) {                              // distribute remainder by desc weight
            int best = -1; float bw = NEG;
            for (int j = 0; j < no; ++j)
                if (!taken[j] && w[j] > bw) { bw = w[j]; best = j; }
            if (best < 0) break;
            alloc[best]++; taken[best] = true; rem--;
        }
        for (int j = 0; j < no; ++j)
            if (alloc[j] > 0) kper[other[j]] = alloc[j] < NPATCH ? alloc[j] : NPATCH;
    }
    // flat candidate lists (layer order, each layer's top-k desc)
    float sc[160]; int lif[160]; int idxf[160];
    float buf[NPATCH];
    int M = 0;
    for (int l = 0; l < LNUM; ++l) {
        int k = kper[l] < NPATCH ? kper[l] : NPATCH;
        if (k <= 0) continue;
        const float* row = z + (size_t)l * NPATCH;
        for (int n = 0; n < NPATCH; ++n) buf[n] = row[n];
        for (int t = 0; t < k && M < 160; ++t) {
            int am = 0; float bv = buf[0];
            for (int n = 1; n < NPATCH; ++n)
                if (buf[n] > bv) { bv = buf[n]; am = n; }
            lif[M] = l; idxf[M] = am; sc[M] = bv; buf[am] = NEG; ++M;
        }
    }
    int final_k   = M < 64 ? M : 64;
    int focus_min = final_k < 32 ? final_k : 32;
    bool used[160];
    for (int i = 0; i < M; ++i) used[i] = false;
    int sel[64]; int ns = 0;
    for (int t = 0; t < focus_min; ++t) {              // focus picks first
        int best = -1; float bv = NEG;
        for (int i = 0; i < M; ++i)
            if (!used[i] && lif[i] == FOCUS && sc[i] > bv) { bv = sc[i]; best = i; }
        if (best < 0) break;
        used[best] = true; sel[ns++] = best;
    }
    int remain = final_k - ns;
    for (int t = 0; t < remain; ++t) {                 // fill with global best
        int best = -1; float bv = NEG;
        for (int i = 0; i < M; ++i)
            if (!used[i] && sc[i] > bv) { bv = sc[i]; best = i; }
        if (best < 0) break;
        used[best] = true; sel[ns++] = best;
    }
    for (int a = 0; a < ns; ++a) {                     // final desc sort by score
        int bsel = a;
        for (int b = a + 1; b < ns; ++b)
            if (sc[sel[b]] > sc[sel[bsel]]) bsel = b;
        int tmp = sel[a]; sel[a] = sel[bsel]; sel[bsel] = tmp;
    }
    for (int i = 0; i < ns; ++i) { sel_li[i] = lif[sel[i]]; sel_idx[i] = idxf[sel[i]]; }
    for (int i = ns; i < 64; ++i) { sel_li[i] = FOCUS; sel_idx[i] = 0; }
}

// out[b,:] = proj[sel_li[b], sel_idx[b], :]  (float4-wide)
__global__ void gather_kernel(const float* __restrict__ proj,
                              const int* __restrict__ sel_li, const int* __restrict__ sel_idx,
                              float* __restrict__ out) {
    int b = blockIdx.x;
    const float4* src = (const float4*)(proj + ((size_t)sel_li[b] * NPATCH + (size_t)sel_idx[b]) * DDIM);
    float4* dst = (float4*)(out + (size_t)b * DDIM);
    for (int i = threadIdx.x; i < DDIM / 4; i += 256) dst[i] = src[i];
}

// ---------------------------------------------------------------------------
extern "C" void kernel_launch(void* const* d_in, const int* in_sizes, int n_in,
                              void* d_out, int out_size, void* d_ws, size_t ws_size,
                              hipStream_t stream) {
    (void)in_sizes; (void)n_in; (void)out_size; (void)ws_size;
    const float* text = (const float*)d_in[0];
    const float* proj = (const float*)d_in[1];
    const float* Wq   = (const float*)d_in[2];
    const float* Wk   = (const float*)d_in[3];
    const float* lnw  = (const float*)d_in[4];
    const float* lnb  = (const float*)d_in[5];
    const float* W1w  = (const float*)d_in[6];
    const float* W1b  = (const float*)d_in[7];
    const float* Wcw  = (const float*)d_in[8];
    const float* Wcb  = (const float*)d_in[9];
    const float* Wiw  = (const float*)d_in[10];
    const float* Wib  = (const float*)d_in[11];
    const float* Wfw  = (const float*)d_in[12];
    const float* Wfb  = (const float*)d_in[13];
    const float* bcv  = (const float*)d_in[14];
    const float* biv  = (const float*)d_in[15];
    const float* bfv  = (const float*)d_in[16];

    float* ws     = (float*)d_ws;
    float* tg     = ws;                    // 4096
    float* y      = ws + 4096;             // 24*4096
    float* c      = ws + 102400;           // 4096
    float* cf     = ws + 106496;           // 4096
    float* s      = ws + 110592;           // 1024
    float* qraw   = ws + 111616;           // 4096
    float* q      = ws + 115712;           // 4096
    float* wq     = ws + 119808;           // 4096
    float* consts = ws + 123904;           // 2 (+pad)
    float* scores = ws + 123908;           // 13824
    float* zsc    = scores + PTOT;         // 13824
    int*   selbuf = (int*)(zsc + PTOT);    // 64 li + 64 idx

    zero_kernel<<<16, 256, 0, stream>>>(c, DDIM);
    text_mean_ln_kernel<<<1, 1024, 0, stream>>>(text, tg);
    layer_mean_kernel<<<(LNUM * (DDIM / 4)) / 256, 256, 0, stream>>>(proj, y);

    for (int l = 0; l < LNUM; ++l) {
        dsu_s_kernel<<<REDDIM, 128, 0, stream>>>(W1w, W1b, c, y + (size_t)l * DDIM, tg, s);
        dsu_c_kernel<<<DDIM, 128, 0, stream>>>(Wcw, Wcb, bcv, Wiw, Wib, biv,
                                               Wfw, Wfb, bfv, s, c, cf, (l == FOCUS) ? 1 : 0);
    }

    qmatvec_kernel<<<DDIM, 256, 0, stream>>>(Wq, cf, qraw);
    q_ln_kernel<<<1, 1024, 0, stream>>>(qraw, lnw, lnb, q, wq, consts);

    score_wmma_kernel<<<PTOT / 64, 256, 0, stream>>>(proj, Wk, wq, consts, scores);

    znorm_kernel<<<LNUM, 256, 0, stream>>>(scores, zsc);
    select_kernel<<<1, 32, 0, stream>>>(zsc, selbuf, selbuf + 64);
    gather_kernel<<<64, 256, 0, stream>>>(proj, selbuf, selbuf + 64, (float*)d_out);
}
